// Unet_Up_69526930588489
// MI455X (gfx1250) — matrix operand order repaired and checked by reading, not compiled, not run on hardware
//
#include <hip/hip_runtime.h>
#include <hip/hip_bf16.h>

// ---------------------------------------------------------------------------
// Types for WMMA fragments (CDNA5 / gfx1250, wave32)
// ---------------------------------------------------------------------------
typedef __bf16  bf16x16 __attribute__((ext_vector_type(16)));
typedef float   v8f     __attribute__((ext_vector_type(8)));
typedef unsigned int u32x4 __attribute__((ext_vector_type(4)));

union FragU {
    bf16x16 v;
    u32x4   u[2];
};

__device__ __forceinline__ unsigned short f2bf(float f) {
    unsigned int u = __float_as_uint(f);
    unsigned int r = u + 0x7FFFu + ((u >> 16) & 1u);   // round-to-nearest-even
    return (unsigned short)(r >> 16);
}

// ---------------------------------------------------------------------------
// Prep kernels
// ---------------------------------------------------------------------------

// fp32 -> bf16 bulk convert
__global__ void cvt_bf16_kernel(const float* __restrict__ X,
                                unsigned short* __restrict__ out, int n) {
    int i = blockIdx.x * blockDim.x + threadIdx.x;
    if (i < n) out[i] = f2bf(X[i]);
}

// Pack weight matrices (nmat x Kdim x Ncols, row-major fp32) into per-lane
// WMMA B fragments (bf16). Tile = (mat, kc, nt); within a tile each lane
// holds 16 bf16: column n = nt*16 + (lane&15), K = kc*32 + (lane>>4)*8 + j (j<8), kc*32 + 16 + (lane>>4)*8 + (j-8) (j>=8).
__global__ void pack_B_kernel(const float* __restrict__ W,
                              unsigned short* __restrict__ P,
                              int nmat, int Kdim, int Ncols) {
    int KC = Kdim >> 5, NT = Ncols >> 4;
    int total = nmat * KC * NT * 32;
    int tid = blockIdx.x * blockDim.x + threadIdx.x;
    if (tid >= total) return;
    int lane = tid & 31;
    int tile = tid >> 5;
    int nt  = tile % NT;
    int tmp = tile / NT;
    int kc  = tmp % KC;
    int mat = tmp / KC;
    int n  = nt * 16 + (lane & 15);
    int hi = lane >> 4;
    unsigned short* dst = P + (size_t)tile * 512 + (size_t)lane * 16;
    for (int j = 0; j < 16; ++j) {
        int K = kc * 32 + hi * 8 + (j < 8 ? j : 16 + (j - 8));
        dst[j] = f2bf(W[((size_t)mat * Kdim + K) * Ncols + n]);
    }
}

// ---------------------------------------------------------------------------
// Transposed conv (parent gather, koff-selected weight, 128 -> 128) via WMMA
// One wave per 16-row tile; 8 masked passes over the 8 weight matrices.
// ---------------------------------------------------------------------------
__global__ void tconv_wmma_kernel(const unsigned short* __restrict__ Xin,  // NIN x 128 bf16
                                  const int* __restrict__ parent,
                                  const int* __restrict__ koff,
                                  const unsigned short* __restrict__ Bp,   // packed 8 x (128x128)
                                  float* __restrict__ out,                 // N x 128 f32
                                  int N) {
    int r0 = blockIdx.x * 16;
    int l  = threadIdx.x;
    int m  = l & 15;
    int hi = l >> 4;
    int r  = r0 + m;

    int p  = parent[r];
    int kk = koff[r];
    const unsigned short* rowp = Xin + (size_t)p * 128;

    v8f acc[8] = {};
    const u32x4 z = {0u, 0u, 0u, 0u};

    for (int k = 0; k < 8; ++k) {
        bool ok = (kk == k);
        for (int kc = 0; kc < 4; ++kc) {
            FragU a;
            int k0 = kc * 32 + hi * 8;
            a.u[0] = ok ? *(const u32x4*)(rowp + k0)      : z;
            a.u[1] = ok ? *(const u32x4*)(rowp + k0 + 16) : z;
            const unsigned short* bt = Bp + (size_t)((k * 4 + kc) * 8) * 512;
            for (int nt = 0; nt < 8; ++nt) {
                FragU b;
                const unsigned short* bl = bt + (size_t)nt * 512 + (size_t)l * 16;
                b.u[0] = *(const u32x4*)(bl);
                b.u[1] = *(const u32x4*)(bl + 8);
                acc[nt] = __builtin_amdgcn_wmma_f32_16x16x32_bf16(
                    false, a.v, false, b.v, (short)0, acc[nt], false, false);
            }
        }
    }
    for (int nt = 0; nt < 8; ++nt)
        for (int i = 0; i < 8; ++i)
            out[(size_t)(r0 + hi * 8 + i) * 128 + nt * 16 + m] = acc[nt][i];
}

// ---------------------------------------------------------------------------
// Generic sparse conv / GEMM via WMMA:  out[N x 64] = sum_k gather_k(X) @ W[k]
// nbr == nullptr -> plain GEMM (identity gather), noff must be 1.
// Kdim in {64, 192}; one wave per 16-row tile, 4 column tiles (C_OUT = 64).
// ---------------------------------------------------------------------------
__global__ void sconv_wmma_kernel(const unsigned short* __restrict__ X,   // ? x Kdim bf16
                                  const unsigned short* __restrict__ Bp,  // packed noff x (Kdim x 64)
                                  const int* __restrict__ nbr,            // noff x N (or null)
                                  const unsigned char* __restrict__ valid,// noff x N (or null)
                                  int noff, int N, int Kdim,
                                  float* __restrict__ out) {              // N x 64 f32
    int r0 = blockIdx.x * 16;
    int l  = threadIdx.x;
    int m  = l & 15;
    int hi = l >> 4;
    int r  = r0 + m;
    int KC = Kdim >> 5;

    v8f acc[4] = {};
    const u32x4 z = {0u, 0u, 0u, 0u};

    for (int k = 0; k < noff; ++k) {
        bool ok;
        size_t idx;
        if (nbr) {
            size_t o = (size_t)k * N + r;
            ok  = valid[o] != 0;
            idx = ok ? (size_t)nbr[o] : 0;
        } else {
            ok  = true;
            idx = (size_t)r;
        }
        const unsigned short* rowp = X + idx * (size_t)Kdim;
        for (int kc = 0; kc < KC; ++kc) {
            FragU a;
            int k0 = kc * 32 + hi * 8;
            a.u[0] = ok ? *(const u32x4*)(rowp + k0)      : z;
            a.u[1] = ok ? *(const u32x4*)(rowp + k0 + 16) : z;
            const unsigned short* bt = Bp + (size_t)((k * KC + kc) * 4) * 512;
            for (int nt = 0; nt < 4; ++nt) {
                FragU b;
                const unsigned short* bl = bt + (size_t)nt * 512 + (size_t)l * 16;
                b.u[0] = *(const u32x4*)(bl);
                b.u[1] = *(const u32x4*)(bl + 8);
                acc[nt] = __builtin_amdgcn_wmma_f32_16x16x32_bf16(
                    false, a.v, false, b.v, (short)0, acc[nt], false, false);
            }
        }
    }
    for (int nt = 0; nt < 4; ++nt)
        for (int i = 0; i < 8; ++i)
            out[(size_t)(r0 + hi * 8 + i) * 64 + nt * 16 + m] = acc[nt][i];
}

// ---------------------------------------------------------------------------
// BatchNorm helpers
// ---------------------------------------------------------------------------

// Per-channel sum / sumsq over N rows. One block (256 thr) per channel.
__global__ void colstats_kernel(const float* __restrict__ X, int N, int C,
                                float* __restrict__ sums) {
    __shared__ float s1[256];
    __shared__ float s2[256];
    int c = blockIdx.x;
    int t = threadIdx.x;
    float a = 0.f, b = 0.f;
    for (int r = t; r < N; r += 256) {
        float v = X[(size_t)r * C + c];
        a += v;
        b += v * v;
    }
    s1[t] = a; s2[t] = b;
    __syncthreads();
    for (int off = 128; off > 0; off >>= 1) {
        if (t < off) { s1[t] += s1[t + off]; s2[t] += s2[t + off]; }
        __syncthreads();
    }
    if (t == 0) { sums[c] = s1[0]; sums[C + c] = s2[0]; }
}

// scale = g * rsqrt(var + eps); shift = b - mean*scale
__global__ void bn_finalize_kernel(const float* __restrict__ sums,
                                   const float* __restrict__ g,
                                   const float* __restrict__ b,
                                   int C, float invN,
                                   float* __restrict__ ss) {
    int c = threadIdx.x;
    if (c >= C) return;
    float mean = sums[c] * invN;
    float var  = sums[C + c] * invN - mean * mean;
    float sc   = g[c] * rsqrtf(var + 1e-5f);
    ss[c]     = sc;
    ss[C + c] = b[c] - mean * sc;
}

// x1[N x 192] bf16 = concat( relu(bn(y0[N x 128])), inter[N x 64] )
__global__ void build_x1_kernel(const float* __restrict__ y0,
                                const float* __restrict__ ss,
                                const float* __restrict__ inter,
                                unsigned short* __restrict__ x1, int N) {
    int i = blockIdx.x * blockDim.x + threadIdx.x;
    if (i >= N * 192) return;
    int row = i / 192, col = i % 192;
    float v;
    if (col < 128) {
        v = y0[(size_t)row * 128 + col] * ss[col] + ss[128 + col];
        v = v > 0.f ? v : 0.f;
    } else {
        v = inter[(size_t)row * 64 + (col - 128)];
    }
    x1[i] = f2bf(v);
}

// out bf16 = relu(bn(X))   (N x C)
__global__ void bn_relu_bf16_kernel(const float* __restrict__ X,
                                    const float* __restrict__ ss,
                                    int N, int C,
                                    unsigned short* __restrict__ out) {
    int i = blockIdx.x * blockDim.x + threadIdx.x;
    if (i >= N * C) return;
    int c = i % C;
    float v = X[i] * ss[c] + ss[C + c];
    v = v > 0.f ? v : 0.f;
    out[i] = f2bf(v);
}

// y = relu(bn(h) + bn(r)); optionally store bf16 (next block input) and/or f32
__global__ void combine_kernel(const float* __restrict__ h,  const float* __restrict__ ssh,
                               const float* __restrict__ r,  const float* __restrict__ ssr,
                               int N, int C,
                               unsigned short* __restrict__ xbf,
                               float* __restrict__ fout) {
    int i = blockIdx.x * blockDim.x + threadIdx.x;
    if (i >= N * C) return;
    int c = i % C;
    float v = h[i] * ssh[c] + ssh[C + c] + r[i] * ssr[c] + ssr[C + c];
    v = v > 0.f ? v : 0.f;
    if (xbf)  xbf[i]  = f2bf(v);
    if (fout) fout[i] = v;
}

// ---------------------------------------------------------------------------
// Host launch
// ---------------------------------------------------------------------------
extern "C" void kernel_launch(void* const* d_in, const int* in_sizes, int n_in,
                              void* d_out, int out_size, void* d_ws, size_t ws_size,
                              hipStream_t stream) {
    (void)in_sizes; (void)n_in; (void)out_size; (void)ws_size;

    constexpr int N    = 60000;
    constexpr int NIN  = 16000;

    const float* x_in   = (const float*)d_in[0];
    const float* inter  = (const float*)d_in[1];
    const int*   parent = (const int*)d_in[2];
    const int*   koff   = (const int*)d_in[3];
    const int*   nbr    = (const int*)d_in[4];
    const unsigned char* valid = (const unsigned char*)d_in[5];
    const float* Wt     = (const float*)d_in[6];
    const float* tg     = (const float*)d_in[7];
    const float* tb     = (const float*)d_in[8];
    const float* b1_W1  = (const float*)d_in[9];
    const float* b1_g1  = (const float*)d_in[10];
    const float* b1_b1  = (const float*)d_in[11];
    const float* b1_W2  = (const float*)d_in[12];
    const float* b1_g2  = (const float*)d_in[13];
    const float* b1_b2  = (const float*)d_in[14];
    const float* b1_Wr  = (const float*)d_in[15];
    const float* b1_gr  = (const float*)d_in[16];
    const float* b1_br  = (const float*)d_in[17];
    const float* b2_W1  = (const float*)d_in[18];
    const float* b2_g1  = (const float*)d_in[19];
    const float* b2_b1  = (const float*)d_in[20];
    const float* b2_W2  = (const float*)d_in[21];
    const float* b2_g2  = (const float*)d_in[22];
    const float* b2_b2  = (const float*)d_in[23];
    const float* b2_Wr  = (const float*)d_in[24];
    const float* b2_gr  = (const float*)d_in[25];
    const float* b2_br  = (const float*)d_in[26];

    // --- workspace bump allocator (256B aligned) ---
    char* wp = (char*)d_ws;
    auto alloc = [&](size_t bytes) -> void* {
        void* p = (void*)wp;
        wp += (bytes + 255) & ~(size_t)255;
        return p;
    };

    unsigned short* xin_bf = (unsigned short*)alloc((size_t)NIN * 128 * 2);
    unsigned short* Wt_p   = (unsigned short*)alloc((size_t)8  * 128 * 128 * 2);
    unsigned short* W1a_p  = (unsigned short*)alloc((size_t)27 * 192 * 64 * 2);
    unsigned short* W2a_p  = (unsigned short*)alloc((size_t)27 * 64  * 64 * 2);
    unsigned short* Wra_p  = (unsigned short*)alloc((size_t)192 * 64 * 2);
    unsigned short* W1b_p  = (unsigned short*)alloc((size_t)27 * 64 * 64 * 2);
    unsigned short* W2b_p  = (unsigned short*)alloc((size_t)27 * 64 * 64 * 2);
    unsigned short* Wrb_p  = (unsigned short*)alloc((size_t)64 * 64 * 2);

    float* y0 = (float*)alloc((size_t)N * 128 * 4);     // reused: h1raw | h2raw
    float* h1raw = y0;                                  // N x 64
    float* h2raw = y0 + (size_t)N * 64;                 // N x 64
    unsigned short* x1_bf = (unsigned short*)alloc((size_t)N * 192 * 2);
    unsigned short* h_bf  = (unsigned short*)alloc((size_t)N * 64 * 2);
    float* rraw = (float*)alloc((size_t)N * 64 * 4);
    unsigned short* x2_bf = (unsigned short*)alloc((size_t)N * 64 * 2);

    float* sums128 = (float*)alloc(256 * 4);
    float* ss128   = (float*)alloc(256 * 4);
    float* sumsA   = (float*)alloc(128 * 4);
    float* ssA     = (float*)alloc(128 * 4);
    float* sumsB   = (float*)alloc(128 * 4);
    float* ssB     = (float*)alloc(128 * 4);
    float* sumsC   = (float*)alloc(128 * 4);
    float* ssC     = (float*)alloc(128 * 4);

    const int rowTiles = N / 16;          // 3750
    const float invN = 1.0f / (float)N;

    auto packGrid = [](int nmat, int K, int C) {
        int total = nmat * (K / 32) * (C / 16) * 32;
        return (total + 127) / 128;
    };

    // --- prep: convert x_in to bf16, pack all weights ---
    cvt_bf16_kernel<<<(NIN * 128 + 255) / 256, 256, 0, stream>>>(x_in, xin_bf, NIN * 128);
    pack_B_kernel<<<packGrid(8, 128, 128), 128, 0, stream>>>(Wt,    Wt_p,  8, 128, 128);
    pack_B_kernel<<<packGrid(27, 192, 64), 128, 0, stream>>>(b1_W1, W1a_p, 27, 192, 64);
    pack_B_kernel<<<packGrid(27, 64, 64),  128, 0, stream>>>(b1_W2, W2a_p, 27, 64, 64);
    pack_B_kernel<<<packGrid(1, 192, 64),  128, 0, stream>>>(b1_Wr, Wra_p, 1, 192, 64);
    pack_B_kernel<<<packGrid(27, 64, 64),  128, 0, stream>>>(b2_W1, W1b_p, 27, 64, 64);
    pack_B_kernel<<<packGrid(27, 64, 64),  128, 0, stream>>>(b2_W2, W2b_p, 27, 64, 64);
    pack_B_kernel<<<packGrid(1, 64, 64),   128, 0, stream>>>(b2_Wr, Wrb_p, 1, 64, 64);

    // --- transposed conv + BN + ReLU + concat -> x1 (N x 192 bf16) ---
    tconv_wmma_kernel<<<rowTiles, 32, 0, stream>>>(xin_bf, parent, koff, Wt_p, y0, N);
    colstats_kernel<<<128, 256, 0, stream>>>(y0, N, 128, sums128);
    bn_finalize_kernel<<<1, 128, 0, stream>>>(sums128, tg, tb, 128, invN, ss128);
    build_x1_kernel<<<(N * 192 + 255) / 256, 256, 0, stream>>>(y0, ss128, inter, x1_bf, N);

    // --- resblock 1 ---
    sconv_wmma_kernel<<<rowTiles, 32, 0, stream>>>(x1_bf, W1a_p, nbr, valid, 27, N, 192, h1raw);
    colstats_kernel<<<64, 256, 0, stream>>>(h1raw, N, 64, sumsA);
    bn_finalize_kernel<<<1, 64, 0, stream>>>(sumsA, b1_g1, b1_b1, 64, invN, ssA);
    bn_relu_bf16_kernel<<<(N * 64 + 255) / 256, 256, 0, stream>>>(h1raw, ssA, N, 64, h_bf);

    sconv_wmma_kernel<<<rowTiles, 32, 0, stream>>>(h_bf, W2a_p, nbr, valid, 27, N, 64, h2raw);
    colstats_kernel<<<64, 256, 0, stream>>>(h2raw, N, 64, sumsB);
    bn_finalize_kernel<<<1, 64, 0, stream>>>(sumsB, b1_g2, b1_b2, 64, invN, ssB);

    sconv_wmma_kernel<<<rowTiles, 32, 0, stream>>>(x1_bf, Wra_p, nullptr, nullptr, 1, N, 192, rraw);
    colstats_kernel<<<64, 256, 0, stream>>>(rraw, N, 64, sumsC);
    bn_finalize_kernel<<<1, 64, 0, stream>>>(sumsC, b1_gr, b1_br, 64, invN, ssC);

    combine_kernel<<<(N * 64 + 255) / 256, 256, 0, stream>>>(h2raw, ssB, rraw, ssC, N, 64,
                                                             x2_bf, nullptr);

    // --- resblock 2 ---
    sconv_wmma_kernel<<<rowTiles, 32, 0, stream>>>(x2_bf, W1b_p, nbr, valid, 27, N, 64, h1raw);
    colstats_kernel<<<64, 256, 0, stream>>>(h1raw, N, 64, sumsA);
    bn_finalize_kernel<<<1, 64, 0, stream>>>(sumsA, b2_g1, b2_b1, 64, invN, ssA);
    bn_relu_bf16_kernel<<<(N * 64 + 255) / 256, 256, 0, stream>>>(h1raw, ssA, N, 64, h_bf);

    sconv_wmma_kernel<<<rowTiles, 32, 0, stream>>>(h_bf, W2b_p, nbr, valid, 27, N, 64, h2raw);
    colstats_kernel<<<64, 256, 0, stream>>>(h2raw, N, 64, sumsB);
    bn_finalize_kernel<<<1, 64, 0, stream>>>(sumsB, b2_g2, b2_b2, 64, invN, ssB);

    sconv_wmma_kernel<<<rowTiles, 32, 0, stream>>>(x2_bf, Wrb_p, nullptr, nullptr, 1, N, 64, rraw);
    colstats_kernel<<<64, 256, 0, stream>>>(rraw, N, 64, sumsC);
    bn_finalize_kernel<<<1, 64, 0, stream>>>(sumsC, b2_gr, b2_br, 64, invN, ssC);

    combine_kernel<<<(N * 64 + 255) / 256, 256, 0, stream>>>(h2raw, ssB, rraw, ssC, N, 64,
                                                             nullptr, (float*)d_out);
}
